// Model_42417097016196
// MI455X (gfx1250) — compile-verified, matrix operand order checked by reading, not run on hardware
//
#include <hip/hip_runtime.h>
#include <hip/hip_bf16.h>

// ---------------- problem dims (fixed by reference) ----------------
#define S_LEN 2048
#define B_SZ  16
#define H_SZ  1024
#define M_SZ  (S_LEN * B_SZ)   // 32768 rows of X
#define N_SZ  (3 * H_SZ)       // 3072 output features per layer
#define K_SZ  H_SZ             // 1024 reduction dim

// ---------------- vector types ----------------
typedef __attribute__((ext_vector_type(16))) __bf16 v16bf;
typedef __attribute__((ext_vector_type(8)))  __bf16 v8bf;
typedef __attribute__((ext_vector_type(8)))  float  v8f;

typedef int i32x4 __attribute__((vector_size(16)));                 // gcc-style int4
typedef __attribute__((address_space(1))) i32x4 gas_i32x4;          // global AS
typedef __attribute__((address_space(3))) i32x4 las_i32x4;          // LDS AS

__device__ __forceinline__ __bf16 f2bf(float f) { return (__bf16)f; }

// ---------------- CDNA5 async global->LDS path (guarded) ----------------
#if __has_builtin(__builtin_amdgcn_global_load_async_to_lds_b128)
#define HAVE_ASYNC_LDS 1
#define ASYNC_CP16(gsrc, ldst)                                                  \
  __builtin_amdgcn_global_load_async_to_lds_b128(                               \
      (gas_i32x4*)(gsrc), (las_i32x4*)(ldst), 0, 0)
#else
#define HAVE_ASYNC_LDS 0
#define ASYNC_CP16(gsrc, ldst)                                                  \
  do { *(uint4*)(ldst) = *(const uint4*)(gsrc); } while (0)
#endif

__device__ __forceinline__ void wait_async0() {
#if __has_builtin(__builtin_amdgcn_s_wait_asynccnt)
  __builtin_amdgcn_s_wait_asynccnt(0);
#else
  asm volatile("s_wait_asynccnt 0x0" ::: "memory");
#endif
}

// =====================================================================
// 1) Embedding gather: X[s,b,h] = emb[x[s,b], h]  (fp32 -> bf16)
// =====================================================================
__global__ void embed_kernel(const int* __restrict__ x,
                             const float* __restrict__ emb,
                             __bf16* __restrict__ X) {
  int gid = blockIdx.x * blockDim.x + threadIdx.x;  // [0, M*H)
  int sb  = gid >> 10;                              // / H (H=1024)
  int h   = gid & 1023;
  int row = x[sb];
  X[gid]  = f2bf(emb[(size_t)row * H_SZ + h]);
}

// =====================================================================
// 2) Weight conversion fp32 -> bf16, all three layers packed
// =====================================================================
__global__ void convw_kernel(const float* __restrict__ W0,
                             const float* __restrict__ W1,
                             const float* __restrict__ W2,
                             __bf16* __restrict__ Wb) {
  int gid = blockIdx.x * blockDim.x + threadIdx.x;  // [0, 3*N*K)
  const int per = N_SZ * K_SZ;
  int layer = gid / per;
  int idx   = gid - layer * per;
  float v = (layer == 0) ? W0[idx] : (layer == 1) ? W1[idx] : W2[idx];
  Wb[gid] = f2bf(v);
}

// =====================================================================
// 3) bf16 WMMA GEMM: Y[m,n] = sum_k A[m,k] * W[n,k] + bias[n]
//    Block tile 128(M) x 256(N), BK=32, 8 waves each computing 64x64
//    (4x4 WMMA tiles) => 16 v_wmma per K-step per wave.
//    Double-buffered LDS fed by async global->LDS copies.
// =====================================================================
#define BM  128
#define BN  256
#define BK  32
#define LDA 40   // padded LDS row stride (elements) to break bank conflicts

__global__ __launch_bounds__(256)
void gemm_kernel(const __bf16* __restrict__ A,
                 const __bf16* __restrict__ Wb,
                 const float* __restrict__ bias,
                 float* __restrict__ Y) {
  __shared__ __attribute__((aligned(16))) __bf16 sA[2][BM * LDA];  // 2 x 10 KB
  __shared__ __attribute__((aligned(16))) __bf16 sB[2][BN * LDA];  // 2 x 20 KB

  const int tid  = threadIdx.x;
  const int lane = tid & 31;
  const int wave = tid >> 5;

  const int m0 = blockIdx.x * BM;           // 256 tiles in M
  const int n0 = blockIdx.y * BN;           // 12 tiles in N
  const int wm = (wave & 1) * 64;           // wave M offset in block tile
  const int wn = (wave >> 1) * 64;          // wave N offset in block tile

  // staging map: thread t moves 8 bf16 (b128) per chunk
  const int ar = tid >> 2;                  // 0..63   (row within 64-row chunk)
  const int ac = (tid & 3) << 3;            // 0,8,16,24 (col in elements)

  // fragment addressing per CDNA5 16-bit A/B layout:
  // lanes 0-15 hold K 0..7 / 16..23, lanes 16-31 hold K 8..15 / 24..31
  const int frow = lane & 15;
  const int kb   = (lane >> 4) << 3;        // 0 or 8

  v8f acc[4][4] = {};

  // ---- prologue: stage 0 loads ----
  {
    const __bf16* ga = A  + (size_t)(m0 + ar) * K_SZ + ac;
    const __bf16* gb = Wb + (size_t)(n0 + ar) * K_SZ + ac;
    ASYNC_CP16(ga,                       &sA[0][ar * LDA + ac]);
    ASYNC_CP16(ga + (size_t)64 * K_SZ,   &sA[0][(ar + 64) * LDA + ac]);
#pragma unroll
    for (int c = 0; c < 4; ++c)
      ASYNC_CP16(gb + (size_t)(c * 64) * K_SZ, &sB[0][(ar + c * 64) * LDA + ac]);
  }

  int stage = 0;
  for (int kk = 0; kk < K_SZ; kk += BK) {
    wait_async0();        // this wave's pending copies into sX[stage] done
    __syncthreads();      // everyone's copies done; prior reads of ^stage done

    // issue copies for the next K-slice into the other buffer (overlaps WMMAs)
    if (kk + BK < K_SZ) {
      const __bf16* ga = A  + (size_t)(m0 + ar) * K_SZ + kk + BK + ac;
      const __bf16* gb = Wb + (size_t)(n0 + ar) * K_SZ + kk + BK + ac;
      ASYNC_CP16(ga,                     &sA[stage ^ 1][ar * LDA + ac]);
      ASYNC_CP16(ga + (size_t)64 * K_SZ, &sA[stage ^ 1][(ar + 64) * LDA + ac]);
#pragma unroll
      for (int c = 0; c < 4; ++c)
        ASYNC_CP16(gb + (size_t)(c * 64) * K_SZ,
                   &sB[stage ^ 1][(ar + c * 64) * LDA + ac]);
    }

    // fragments + 16 WMMAs from current stage
    v16bf af[4], bfrag[4];
#pragma unroll
    for (int i = 0; i < 4; ++i) {
      const __bf16* p = &sA[stage][(wm + i * 16 + frow) * LDA + kb];
      v8bf lo = *(const v8bf*)p;
      v8bf hi = *(const v8bf*)(p + 16);
      af[i] = __builtin_shufflevector(lo, hi, 0,1,2,3,4,5,6,7,8,9,10,11,12,13,14,15);
    }
#pragma unroll
    for (int j = 0; j < 4; ++j) {
      const __bf16* p = &sB[stage][(wn + j * 16 + frow) * LDA + kb];
      v8bf lo = *(const v8bf*)p;
      v8bf hi = *(const v8bf*)(p + 16);
      bfrag[j] = __builtin_shufflevector(lo, hi, 0,1,2,3,4,5,6,7,8,9,10,11,12,13,14,15);
    }

#pragma unroll
    for (int i = 0; i < 4; ++i)
#pragma unroll
      for (int j = 0; j < 4; ++j)
        acc[i][j] = __builtin_amdgcn_wmma_f32_16x16x32_bf16(
            /*neg_a=*/false, af[i], /*neg_b=*/false, bfrag[j],
            /*c_mod=*/(short)0, acc[i][j],
            /*reuse_a=*/false, /*reuse_b=*/false);

    stage ^= 1;
  }

  // epilogue: C/D layout — lane<16: VGPR r = (M=r, N=lane); lane>=16: (M=8+r, N=lane-16)
  const int rbase = (lane >> 4) << 3;
#pragma unroll
  for (int j = 0; j < 4; ++j) {
    const int n = n0 + wn + j * 16 + frow;
    const float bv = bias[n];
#pragma unroll
    for (int i = 0; i < 4; ++i) {
#pragma unroll
      for (int r = 0; r < 8; ++r) {
        const int m = m0 + wm + i * 16 + rbase + r;
        Y[(size_t)m * N_SZ + n] = acc[i][j][r] + bv;
      }
    }
  }
}

// =====================================================================
// 4) Fused gates + forget-mult scan + output gate.
//    One thread per (b,h) channel; sequential over s (true dependency).
//    last==0: write bf16 X for next layer; last==1: write fp32 out[:-1]
// =====================================================================
__global__ void scan_kernel(const float* __restrict__ Y,
                            __bf16* __restrict__ Xn,
                            float* __restrict__ out,
                            int last) {
  int t = blockIdx.x * blockDim.x + threadIdx.x;  // [0, B*H)
  int b = t >> 10;    // / H
  int h = t & 1023;
  float c = 0.0f;
  for (int s = 0; s < S_LEN; ++s) {
    size_t base = ((size_t)s * B_SZ + b) * N_SZ;
    float z = Y[base + h];
    float f = Y[base + H_SZ + h];
    float o = Y[base + 2 * H_SZ + h];
    float sf = 1.0f / (1.0f + __expf(-f));
    float tz = 2.0f / (1.0f + __expf(-2.0f * z)) - 1.0f;   // tanh(z)
    c = sf * tz + (1.0f - sf) * c;
    float so  = 1.0f / (1.0f + __expf(-o));
    float res = so * c;
    size_t oi = ((size_t)s * B_SZ + b) * H_SZ + h;
    if (last) {
      if (s < S_LEN - 1) out[oi] = res;
    } else {
      Xn[oi] = f2bf(res);
    }
  }
}

// =====================================================================
// launcher
// =====================================================================
extern "C" void kernel_launch(void* const* d_in, const int* in_sizes, int n_in,
                              void* d_out, int out_size, void* d_ws, size_t ws_size,
                              hipStream_t stream) {
  const int*   x   = (const int*)d_in[0];
  const float* emb = (const float*)d_in[1];
  const float* W[3]  = {(const float*)d_in[2], (const float*)d_in[4], (const float*)d_in[6]};
  const float* bb[3] = {(const float*)d_in[3], (const float*)d_in[5], (const float*)d_in[7]};

  char* ws = (char*)d_ws;
  __bf16* Xb = (__bf16*)ws;                                          // M*K bf16   (64 MB)
  float*  Yf = (float*)(ws + (size_t)M_SZ * K_SZ * 2);               // M*N f32    (384 MB)
  __bf16* Wb = (__bf16*)(ws + (size_t)M_SZ * K_SZ * 2
                            + (size_t)M_SZ * N_SZ * 4);              // 3*N*K bf16 (19 MB)

  // 1) embedding gather -> bf16 X
  embed_kernel<<<(M_SZ * H_SZ) / 256, 256, 0, stream>>>(x, emb, Xb);

  // 2) weights fp32 -> bf16
  convw_kernel<<<(3 * N_SZ * K_SZ) / 256, 256, 0, stream>>>(W[0], W[1], W[2], Wb);

  // 3) three layers: WMMA GEMM + fused scan
  dim3 ggrid(M_SZ / BM, N_SZ / BN);   // 256 x 12
  for (int l = 0; l < 3; ++l) {
    gemm_kernel<<<ggrid, 256, 0, stream>>>(Xb, Wb + (size_t)l * N_SZ * K_SZ, bb[l], Yf);
    scan_kernel<<<(B_SZ * H_SZ) / 256, 256, 0, stream>>>(
        Yf, Xb, (float*)d_out, (l == 2) ? 1 : 0);
  }
}
